// Encoder_47553877901790
// MI455X (gfx1250) — matrix operand order repaired and checked by reading, not compiled, not run on hardware
//
#include <hip/hip_runtime.h>

// ---------------------------------------------------------------------------
// Transformer encoder forward for gfx1250 (MI455X), bf16 WMMA everywhere.
// L=6, B=4, T=1024, D=512, H=8, DH=64, F=2048.
// GEMMs + attention: double-buffered async global->LDS staging (ASYNCcnt path).
// ---------------------------------------------------------------------------

typedef __bf16 bf16;
typedef __attribute__((ext_vector_type(8)))  bf16  v8bf;
typedef __attribute__((ext_vector_type(16))) bf16  v16bf;
typedef __attribute__((ext_vector_type(8)))  float v8f;

static constexpr int Bx = 4, Tx = 1024, Dx = 512, Hx = 8, DHx = 64, Fx = 2048, Lx = 6;
static constexpr int MT = Bx * Tx;           // 4096 token rows

// ---- WMMA helpers ----------------------------------------------------------
__device__ inline v8f wmma_bf16(v16bf a, v16bf b, v8f c) {
  // D = A(16x32 bf16) * B(32x16 bf16) + C(16x16 f32)
  return __builtin_amdgcn_wmma_f32_16x16x32_bf16(false, a, false, b, (short)0, c,
                                                 false, false);
}

// A-fragment (16x32, 16-bit): lane m=lane&15 holds row m.
// Contiguous K order (matches B): lane<16 -> elems 0..15 = K 0..15,
//                                 lane>=16 -> elems 0..15 = K 16..31
// caller passes base = row_ptr + kb + 16*half  (one contiguous 32B run)
__device__ inline v16bf load_a_frag(const bf16* base) {
  v8bf lo = *(const v8bf*)(base);
  v8bf hi = *(const v8bf*)(base + 8);
  v16bf r;
#pragma unroll
  for (int i = 0; i < 8; ++i) { r[i] = lo[i]; r[i + 8] = hi[i]; }
  return r;
}
// B-fragment (32x16, 16-bit): lane n=lane&15 holds column n;
// lane<16 -> K 0..15 contiguous, lane>=16 -> K 16..31 contiguous.
// With W stored transposed [N][K], this is one 32-byte contiguous load.

// LDS byte offset of a __shared__ pointer: flat LDS aperture keeps the LDS
// offset in addr[31:0] (ISA 10.2 aperture table), so truncation is exact.
__device__ inline unsigned lds_off(const void* p) {
  return (unsigned)(uintptr_t)p;
}

// ---- x = x + pos -----------------------------------------------------------
__global__ void addpos_kernel(const float* __restrict__ x,
                              const float* __restrict__ pos,
                              float* __restrict__ out) {
  int i = blockIdx.x * 256 + threadIdx.x;         // MT*Dx total
  out[i] = x[i] + pos[i & (Tx * Dx - 1)];
}

// ---- f32 [K][N] -> bf16 [N][K] (convert + transpose) -----------------------
__global__ void convT_kernel(const float* __restrict__ in, bf16* __restrict__ out,
                             int K, int N) {
  int i = blockIdx.x * 256 + threadIdx.x;         // K*N total
  int k = i / N, n = i - k * N;
  out[(size_t)n * K + k] = (bf16)in[i];
}

// ---- LayerNorm: one wave per 512-wide row, writes bf16 ---------------------
__global__ __launch_bounds__(256)
void ln_kernel(const float* __restrict__ x, const float* __restrict__ w,
               const float* __restrict__ bia, bf16* __restrict__ out) {
  const int wid = threadIdx.x >> 5, lane = threadIdx.x & 31;
  const int row = blockIdx.x * 8 + wid;           // 4096 rows
  const float* xr = x + (size_t)row * Dx;
  float v[16];
  float s = 0.f;
#pragma unroll
  for (int i = 0; i < 16; ++i) { v[i] = xr[lane + i * 32]; s += v[i]; }
#pragma unroll
  for (int m = 1; m < 32; m <<= 1) s += __shfl_xor(s, m, 32);
  const float mean = s * (1.0f / Dx);
  float var = 0.f;
#pragma unroll
  for (int i = 0; i < 16; ++i) { float d = v[i] - mean; var += d * d; }
#pragma unroll
  for (int m = 1; m < 32; m <<= 1) var += __shfl_xor(var, m, 32);
  const float rstd = rsqrtf(var * (1.0f / Dx) + 1e-5f);
  bf16* orow = out + (size_t)row * Dx;
#pragma unroll
  for (int i = 0; i < 16; ++i) {
    int c = lane + i * 32;
    orow[c] = (bf16)((v[i] - mean) * rstd * w[c] + bia[c]);
  }
}

// ---- GEMM: C[M,N] = A[M,K](bf16) * Wt[N,K](bf16)^T + bias, epilogue modes --
// WG = 8 waves, tile 64(M) x 128(N). B tile (128x32 bf16 = 8KB) is staged in
// LDS with double-buffered GLOBAL_LOAD_ASYNC_TO_LDS_B128 (ASYNCcnt-tracked),
// shared by all 8 waves (4x less L2 traffic on the B side, copy/compute overlap).
enum { EP_BF16 = 0, EP_VT = 1, EP_RESID = 2, EP_RELU = 3 };

template <int MODE>
__global__ __launch_bounds__(256)
void gemm_wmma(const bf16* __restrict__ A, const bf16* __restrict__ Wt,
               const float* __restrict__ bias, const float* resid,
               bf16* __restrict__ outb, float* outf, int M, int N, int K) {
  __shared__ __align__(64) bf16 btile[2][128 * 32];   // [row][kcol], 64B rows
  const int tid  = threadIdx.x;
  const int wid  = tid >> 5;
  const int lane = tid & 31;
  const int half = lane >> 4;
  const int l16  = lane & 15;
  const int wn   = wid >> 2;                          // 0..1 -> +wn*64 in N
  const int m0 = blockIdx.x * 64 + (wid & 3) * 16;    // 4 waves along M
  const int n0 = blockIdx.y * 128 + wn * 64;
  const bf16* arow = A + (size_t)(m0 + l16) * K + 16 * half;

  // this wave's 1KB slice of the 8KB B tile: 2 async b128 ops (16B per lane)
  const int off0 = wid * 1024 + lane * 16;            // byte offset in tile
  const int off1 = off0 + 512;
  const int r0 = off0 >> 6, c0 = off0 & 63;           // tile row / byte-in-row
  const int r1 = off1 >> 6, c1 = off1 & 63;
  const char* wrow0 = (const char*)(Wt + (size_t)(blockIdx.y * 128 + r0) * K) + c0;
  const char* wrow1 = (const char*)(Wt + (size_t)(blockIdx.y * 128 + r1) * K) + c1;
  const unsigned d0a = lds_off(&btile[0][0]) + off0;
  const unsigned d1a = d0a + 512;
  const unsigned dstride = lds_off(&btile[1][0]) - lds_off(&btile[0][0]);

  const v8f z = {0.f, 0.f, 0.f, 0.f, 0.f, 0.f, 0.f, 0.f};
  v8f acc[4] = {z, z, z, z};

  auto issue = [&](int buf, int kb) {
    const char* g0 = wrow0 + (size_t)kb * 2;
    const char* g1 = wrow1 + (size_t)kb * 2;
    unsigned dd0 = d0a + (unsigned)buf * dstride;
    unsigned dd1 = d1a + (unsigned)buf * dstride;
    asm volatile("global_load_async_to_lds_b128 %0, %1, off"
                 :: "v"(dd0), "v"(g0) : "memory");
    asm volatile("global_load_async_to_lds_b128 %0, %1, off"
                 :: "v"(dd1), "v"(g1) : "memory");
  };

  const int nt = K >> 5;                              // K / 32 tiles
  issue(0, 0);
  for (int it = 0; it < nt; ++it) {
    const int kb = it << 5;
    if (it + 1 < nt) {
      issue((it + 1) & 1, kb + 32);
      // the 2 just-issued may remain; previous tile's 2 (in-order) are done
      asm volatile("s_wait_asynccnt 0x2" ::: "memory");
    } else {
      asm volatile("s_wait_asynccnt 0x0" ::: "memory");
    }
    __syncthreads();                                  // tile `it` visible to WG

    v16bf a = load_a_frag(arow + kb);
    const bf16* bt = &btile[it & 1][0];
#pragma unroll
    for (int f = 0; f < 4; ++f) {
      const bf16* bp = bt + ((wn * 64 + f * 16 + l16) * 32 + 16 * half);
      v16bf b = *(const v16bf*)bp;                    // 2x ds_load_b128
      acc[f] = wmma_bf16(a, b, acc[f]);
    }
    __syncthreads();                                  // done reading buf[it&1]
  }

#pragma unroll
  for (int f = 0; f < 4; ++f) {
    const int n = n0 + f * 16 + l16;
    const float bn = bias[n];
#pragma unroll
    for (int r = 0; r < 8; ++r) {
      const int m = m0 + r + 8 * half;               // C layout: M = r + 8*half
      float val = acc[f][r] + bn;
      if constexpr (MODE == EP_BF16) {
        outb[(size_t)m * N + n] = (bf16)val;
      } else if constexpr (MODE == EP_RELU) {
        outb[(size_t)m * N + n] = (bf16)fmaxf(val, 0.f);
      } else if constexpr (MODE == EP_RESID) {
        outf[(size_t)m * N + n] = resid[(size_t)m * N + n] + val;
      } else {  // EP_VT: scatter V as [b][h][dh][t] for the PV B-fragments
        const int bb = m >> 10, t = m & (Tx - 1);
        const int hh = n >> 6,  dh = n & (DHx - 1);
        outb[(((size_t)(bb * Hx + hh) * DHx + dh) << 10) + t] = (bf16)val;
      }
    }
  }
}

// ---- Flash attention: 4 waves share one (b,h); K/V tiles async-staged ------
// Each wave owns 16 query rows. Per 32-key step the block cooperatively stages
// the 32x64 K tile ([j][dh]) and 64x32 V^T tile ([dh][j]) into double-buffered
// LDS via GLOBAL_LOAD_ASYNC_TO_LDS_B128 (4x less L2 traffic, copy/compute overlap).
__global__ __launch_bounds__(128)
void attn_kernel(const bf16* __restrict__ q, const bf16* __restrict__ k,
                 const bf16* __restrict__ vt, bf16* __restrict__ o) {
  __shared__ __align__(64) bf16 ktile[2][32 * 64];  // [j][dh], 128B rows
  __shared__ __align__(64) bf16 vtile[2][64 * 32];  // [dh][j], 64B rows
  __shared__ __align__(64) bf16 plds[4][16 * 32];   // per-wave P staging tile
  const int tid  = threadIdx.x;
  const int wid  = tid >> 5;
  const int lane = tid & 31;
  const int half = lane >> 4;
  const int l16  = lane & 15;
  const int gid = blockIdx.x * 4 + wid;             // 2048 waves total
  const int b  = gid >> 9;                          // 512 waves per batch
  const int h  = (gid >> 6) & (Hx - 1);             // uniform across the block
  const int i0 = (gid & 63) << 4;                   // query block start

  // Q A-fragments (rows i0..i0+15, dh split 0..31 / 32..63)
  const bf16* qbase = q + (size_t)(b * Tx + i0 + l16) * Dx + h * DHx + 16 * half;
  const v16bf qa0 = load_a_frag(qbase);
  const v16bf qa1 = load_a_frag(qbase + 32);

  // async-copy assignment: each thread moves 32B of each 4KB tile
  const int off = tid * 32;                         // byte offset within tile
  const char* kbase = (const char*)(k + (size_t)(b * Tx + (off >> 7)) * Dx + h * DHx)
                      + (off & 127);                // K tile row = off/128
  const char* vbase = (const char*)(vt + (size_t)((b * Hx + h) * DHx + (off >> 6)) * Tx)
                      + (off & 63);                 // V tile row = off/64
  const unsigned kdst = lds_off(&ktile[0][0]) + off;
  const unsigned vdst = lds_off(&vtile[0][0]) + off;
  const unsigned kstride = lds_off(&ktile[1][0]) - lds_off(&ktile[0][0]);
  const unsigned vstride = lds_off(&vtile[1][0]) - lds_off(&vtile[0][0]);

  auto issue = [&](int buf, int jb) {
    const char* gk = kbase + (size_t)jb * (Dx * 2); // advance jb key rows
    const char* gv = vbase + (size_t)jb * 2;        // advance jb key cols
    unsigned dk = kdst + (unsigned)buf * kstride;
    unsigned dv = vdst + (unsigned)buf * vstride;
    asm volatile("global_load_async_to_lds_b128 %0, %1, off"
                 :: "v"(dk), "v"(gk) : "memory");
    asm volatile("global_load_async_to_lds_b128 %0, %1, off"
                 :: "v"(dk + 16u), "v"(gk + 16) : "memory");
    asm volatile("global_load_async_to_lds_b128 %0, %1, off"
                 :: "v"(dv), "v"(gv) : "memory");
    asm volatile("global_load_async_to_lds_b128 %0, %1, off"
                 :: "v"(dv + 16u), "v"(gv + 16) : "memory");
  };

  const v8f z = {0.f, 0.f, 0.f, 0.f, 0.f, 0.f, 0.f, 0.f};
  v8f acc[4] = {z, z, z, z};                        // O accumulator, dh 0..63
  float mrow[8], lrow[8];
#pragma unroll
  for (int r = 0; r < 8; ++r) { mrow[r] = -1e30f; lrow[r] = 0.f; }
  bf16* pl = &plds[wid][0];

  constexpr int NT = Tx / 32;                       // 32 key-tiles
  issue(0, 0);
  for (int it = 0; it < NT; ++it) {
    if (it + 1 < NT) {
      issue((it + 1) & 1, (it + 1) << 5);
      asm volatile("s_wait_asynccnt 0x4" ::: "memory");  // prev tile's 4 done
    } else {
      asm volatile("s_wait_asynccnt 0x0" ::: "memory");
    }
    __syncthreads();                                // tiles visible to all waves

    const bf16* kt  = &ktile[it & 1][0];
    const bf16* vtl = &vtile[it & 1][0];

    // S = Q K^T for 32 keys (two 16-wide j tiles, K-dim = dh in 2 steps)
    const bf16* k0 = kt + (l16 * 64 + 16 * half);
    const bf16* k1 = kt + ((16 + l16) * 64 + 16 * half);
    v8f s0 = z, s1 = z;
    s0 = wmma_bf16(qa0, *(const v16bf*)(k0), s0);
    s0 = wmma_bf16(qa1, *(const v16bf*)(k0 + 32), s0);
    s1 = wmma_bf16(qa0, *(const v16bf*)(k1), s1);
    s1 = wmma_bf16(qa1, *(const v16bf*)(k1 + 32), s1);

    // online softmax update; row r+8*half spans the 16 lanes of this half
#pragma unroll
    for (int r = 0; r < 8; ++r) {
      float a0 = s0[r] * 0.125f;                    // 1/sqrt(DH)
      float a1 = s1[r] * 0.125f;
      float t = fmaxf(a0, a1);
#pragma unroll
      for (int m = 1; m < 16; m <<= 1) t = fmaxf(t, __shfl_xor(t, m, 32));
      const float mn = fmaxf(mrow[r], t);
      const float alpha = __expf(mrow[r] - mn);
      const float p0 = __expf(a0 - mn);
      const float p1 = __expf(a1 - mn);
      float sum = p0 + p1;
#pragma unroll
      for (int m = 1; m < 16; m <<= 1) sum += __shfl_xor(sum, m, 32);
      lrow[r] = lrow[r] * alpha + sum;
      mrow[r] = mn;
      acc[0][r] *= alpha; acc[1][r] *= alpha;
      acc[2][r] *= alpha; acc[3][r] *= alpha;
      // stage P (C layout) into LDS row-major 16x32 tile
      pl[(r + 8 * half) * 32 + l16]      = (bf16)p0;
      pl[(r + 8 * half) * 32 + l16 + 16] = (bf16)p1;
    }
    asm volatile("s_wait_dscnt 0" ::: "memory");    // DS stores -> DS loads (same wave)

    // P as A-fragment, V^T rows as B-fragments: O += P @ V
    const v16bf pa = load_a_frag(pl + l16 * 32 + 16 * half);
#pragma unroll
    for (int f = 0; f < 4; ++f) {
      const v16bf vb = *(const v16bf*)(vtl + (f * 16 + l16) * 32 + 16 * half);
      acc[f] = wmma_bf16(pa, vb, acc[f]);
    }
    __syncthreads();                                // done reading buffers
  }

#pragma unroll
  for (int f = 0; f < 4; ++f) {
#pragma unroll
    for (int r = 0; r < 8; ++r) {
      const int row = b * Tx + i0 + r + 8 * half;
      const int col = h * DHx + f * 16 + l16;
      o[(size_t)row * Dx + col] = (bf16)(acc[f][r] / lrow[r]);
    }
  }
}

// ---------------------------------------------------------------------------
extern "C" void kernel_launch(void* const* d_in, const int* in_sizes, int n_in,
                              void* d_out, int out_size, void* d_ws, size_t ws_size,
                              hipStream_t stream) {
  (void)in_sizes; (void)n_in; (void)out_size; (void)ws_size;
  const float* x_in = (const float*)d_in[0];
  const float* pos  = (const float*)d_in[1];
  const float* ln1w = (const float*)d_in[2];
  const float* ln1b = (const float*)d_in[3];
  const float* wq   = (const float*)d_in[4];
  const float* bq   = (const float*)d_in[5];
  const float* wk   = (const float*)d_in[6];
  const float* bk   = (const float*)d_in[7];
  const float* wv   = (const float*)d_in[8];
  const float* bv   = (const float*)d_in[9];
  const float* wo   = (const float*)d_in[10];
  const float* bo   = (const float*)d_in[11];
  const float* ln2w = (const float*)d_in[12];
  const float* ln2b = (const float*)d_in[13];
  const float* w1   = (const float*)d_in[14];
  const float* b1   = (const float*)d_in[15];
  const float* w2   = (const float*)d_in[16];
  const float* b2   = (const float*)d_in[17];

  // workspace carve-up (~80 MB total)
  char* p = (char*)d_ws;
  auto carve = [&](size_t bytes) {
    char* r = p; p += (bytes + 255) & ~(size_t)255; return r;
  };
  const size_t WSQ = (size_t)Lx * Dx * Dx;      // 512x512 per layer
  const size_t WSF = (size_t)Lx * Dx * Fx;      // 512x2048 per layer
  bf16* wq_t = (bf16*)carve(WSQ * 2);
  bf16* wk_t = (bf16*)carve(WSQ * 2);
  bf16* wv_t = (bf16*)carve(WSQ * 2);
  bf16* wo_t = (bf16*)carve(WSQ * 2);
  bf16* w1_t = (bf16*)carve(WSF * 2);
  bf16* w2_t = (bf16*)carve(WSF * 2);
  float* x    = (float*)carve((size_t)MT * Dx * 4);
  bf16*  h_bf = (bf16*)carve((size_t)MT * Dx * 2);
  bf16*  q_bf = (bf16*)carve((size_t)MT * Dx * 2);
  bf16*  k_bf = (bf16*)carve((size_t)MT * Dx * 2);
  bf16*  vt_bf= (bf16*)carve((size_t)MT * Dx * 2);
  bf16*  o_bf = (bf16*)carve((size_t)MT * Dx * 2);
  bf16*  ff_bf= (bf16*)carve((size_t)MT * Fx * 2);

  // x = x + pos
  addpos_kernel<<<(MT * Dx) / 256, 256, 0, stream>>>(x_in, pos, x);

  // weights -> bf16, transposed [N][K]
  for (int l = 0; l < Lx; ++l) {
    const size_t oq = (size_t)l * Dx * Dx, of = (size_t)l * Dx * Fx;
    convT_kernel<<<(Dx * Dx) / 256, 256, 0, stream>>>(wq + oq, wq_t + oq, Dx, Dx);
    convT_kernel<<<(Dx * Dx) / 256, 256, 0, stream>>>(wk + oq, wk_t + oq, Dx, Dx);
    convT_kernel<<<(Dx * Dx) / 256, 256, 0, stream>>>(wv + oq, wv_t + oq, Dx, Dx);
    convT_kernel<<<(Dx * Dx) / 256, 256, 0, stream>>>(wo + oq, wo_t + oq, Dx, Dx);
    convT_kernel<<<(Dx * Fx) / 256, 256, 0, stream>>>(w1 + of, w1_t + of, Dx, Fx);
    convT_kernel<<<(Fx * Dx) / 256, 256, 0, stream>>>(w2 + of, w2_t + of, Fx, Dx);
  }

  const dim3 gD(MT / 64, Dx / 128);   // N = 512 GEMMs
  const dim3 gF(MT / 64, Fx / 128);   // N = 2048 GEMM

  for (int l = 0; l < Lx; ++l) {
    const size_t oq = (size_t)l * Dx * Dx, of = (size_t)l * Dx * Fx;
    // --- attention block ---
    ln_kernel<<<MT / 8, 256, 0, stream>>>(x, ln1w + l * Dx, ln1b + l * Dx, h_bf);
    gemm_wmma<EP_BF16><<<gD, 256, 0, stream>>>(h_bf, wq_t + oq, bq + l * Dx,
                                               nullptr, q_bf, nullptr, MT, Dx, Dx);
    gemm_wmma<EP_BF16><<<gD, 256, 0, stream>>>(h_bf, wk_t + oq, bk + l * Dx,
                                               nullptr, k_bf, nullptr, MT, Dx, Dx);
    gemm_wmma<EP_VT><<<gD, 256, 0, stream>>>(h_bf, wv_t + oq, bv + l * Dx,
                                             nullptr, vt_bf, nullptr, MT, Dx, Dx);
    attn_kernel<<<(Bx * Hx * (Tx / 16)) / 4, 128, 0, stream>>>(q_bf, k_bf, vt_bf, o_bf);
    gemm_wmma<EP_RESID><<<gD, 256, 0, stream>>>(o_bf, wo_t + oq, bo + l * Dx,
                                                x, nullptr, x, MT, Dx, Dx);
    // --- feed-forward block ---
    ln_kernel<<<MT / 8, 256, 0, stream>>>(x, ln2w + l * Dx, ln2b + l * Dx, h_bf);
    gemm_wmma<EP_RELU><<<gF, 256, 0, stream>>>(h_bf, w1_t + of, b1 + l * Fx,
                                               nullptr, ff_bf, nullptr, MT, Fx, Dx);
    float* xdst = (l == Lx - 1) ? (float*)d_out : x;
    gemm_wmma<EP_RESID><<<gD, 256, 0, stream>>>(ff_bf, w2_t + of, b2 + l * Dx,
                                                x, nullptr, xdst, MT, Dx, Fx);
  }
}
